// ScaledDotProductAttention_29094108463701
// MI455X (gfx1250) — matrix operand order repaired and checked by reading, not compiled, not run on hardware
//
#include <hip/hip_runtime.h>

typedef __attribute__((ext_vector_type(16))) __bf16 v16bf;
typedef __attribute__((ext_vector_type(8)))  float  v8f;
typedef __attribute__((ext_vector_type(4)))  float  f32x4;

#define B_   4
#define H_   16
#define S_   2048
#define D_   128
#define QT   16          // query rows per workgroup
#define NW   8           // waves per workgroup (wave32)
#define TPB  256
#define QROWB 136        // bf16 Q-tile row stride (pad: 272B -> +4 banks/row)
#define SROWF 2052       // f32 logits row stride (pad: 8208B -> +4 banks/row)
#define PROWB 2056       // bf16 P-tile row stride (pad: 4112B -> +4 banks/row)

#define QBYTES (QT * QROWB * 2)
#define SBYTES (QT * SROWF * 4)
#define PBYTES (QT * PROWB * 2)
#define SMEM_BYTES (QBYTES + SBYTES + PBYTES + 64)   // ~197 KB of the 320 KB WGP LDS

__global__ __launch_bounds__(TPB)
void ScaledDotProductAttention_29094108463701_kernel(
    const float* __restrict__ Q, const float* __restrict__ K,
    const float* __restrict__ V, float* __restrict__ Out,
    float* __restrict__ Attw)
{
  extern __shared__ char smem[];
  __bf16* Qbf    = (__bf16*)smem;                              // [16][QROWB]
  float*  Ss     = (float*)(smem + QBYTES);                    // [16][SROWF]
  __bf16* Pbf    = (__bf16*)(smem + QBYTES + SBYTES);          // [16][PROWB]
  float*  rowInv = (float*)(smem + QBYTES + SBYTES + PBYTES);  // [16]

  const int bh = blockIdx.x >> 7;          // S_/QT = 128 q-tiles per (b,h)
  const int qt = blockIdx.x & 127;
  const size_t qkvBase = (size_t)bh * S_ * D_;
  const float* Qb = Q + qkvBase + (size_t)qt * QT * D_;
  const float* Kb = K + qkvBase;
  const float* Vb = V + qkvBase;
  float* Ob = Out + qkvBase + (size_t)qt * QT * D_;
  float* Pg = Attw + (size_t)bh * S_ * S_ + (size_t)qt * QT * S_;

  const int tid  = threadIdx.x;
  const int lane = tid & 31;
  const int wave = tid >> 5;
  const int hi   = lane >> 4;    // which 16-lane half of the wave
  const int l16  = lane & 15;

  // ---- Phase 0: Q tile -> LDS as bf16 (row-major, padded) -------------------
  for (int i = tid; i < QT * D_; i += TPB) {
    const int r = i >> 7, c = i & (D_ - 1);
    Qbf[r * QROWB + c] = (__bf16)Qb[i];
  }
  __syncthreads();

  // ---- Phase 1: logits = (1/sqrt(D)) * Q K^T via v_wmma_f32_16x16x32_bf16 ---
  // A 16x32 bf16: lane(m=l16): elems 0..7 -> K = 32c + 8*hi + e,
  //                            elems 8..15 -> K = 32c + 16 + 8*hi + e
// B 32x16 bf16: lane(n=l16): same K mapping as A
  const float scale = 0.08838834764831843f;   // 1/sqrt(128)
  const __bf16* arow = Qbf + l16 * QROWB;
  for (int j = wave; j < S_ / 16; j += NW) {
    const float* krow = Kb + (size_t)(j * 16 + l16) * D_;
    v8f acc = {};
#pragma unroll
    for (int c = 0; c < D_ / 32; ++c) {
      v16bf a, b;
      const int ka = c * 32 + hi * 8;
#pragma unroll
      for (int e = 0; e < 8; ++e) { a[e] = arow[ka + e]; a[8 + e] = arow[ka + 16 + e]; }
      const int kb = c * 32 + hi * 8;
#pragma unroll
      for (int e = 0; e < 8; ++e) { b[e] = (__bf16)krow[kb + e]; b[8 + e] = (__bf16)krow[kb + 16 + e]; }
      acc = __builtin_amdgcn_wmma_f32_16x16x32_bf16(false, a, false, b,
                                                    (short)0, acc, false, false);
    }
    // C layout: VGPR i, lanes 0-15 -> M=i, lanes 16-31 -> M=8+i; N=l16
#pragma unroll
    for (int i = 0; i < 8; ++i)
      Ss[(i + hi * 8) * SROWF + j * 16 + l16] = acc[i] * scale;
  }
  __syncthreads();

  // ---- Phase 2: per-row softmax stats (16 lanes per row, shfl reductions) ---
  {
    const int r = tid >> 4, sub = tid & 15;
    float* srow = Ss + r * SROWF;
    float m = -3.0e38f;
    for (int c = sub; c < S_; c += 16) m = fmaxf(m, srow[c]);
#pragma unroll
    for (int off = 8; off; off >>= 1) m = fmaxf(m, __shfl_xor(m, off, 16));
    float sum = 0.f;
    for (int c = sub; c < S_; c += 16) {
      const float e = __expf(srow[c] - m);
      srow[c] = e;                       // overwrite logits with exp(s - m)
      sum += e;
    }
#pragma unroll
    for (int off = 8; off; off >>= 1) sum += __shfl_xor(sum, off, 16);
    if (sub == 0) rowInv[r] = 1.0f / sum;
  }
  __syncthreads();

  // ---- Phase 3: normalize -> stream P to HBM (f32, 1 GiB total, NT stores
  //      so the write-once stream does not evict the K/V working set from the
  //      192 MB L2) + bf16 copy of P into LDS for the PV matmul --------------
  for (int i = tid; i < QT * S_ / 4; i += TPB) {
    const int r = i >> 9, c4 = (i & 511) << 2;   // S_/4 = 512 float4 per row
    const float inv = rowInv[r];
    const float* sp = Ss + r * SROWF + c4;
    f32x4 p;
    p.x = sp[0] * inv; p.y = sp[1] * inv; p.z = sp[2] * inv; p.w = sp[3] * inv;
    __builtin_nontemporal_store(p, (f32x4*)(Pg + (size_t)r * S_ + c4)); // b128 NT
    __bf16* pb = Pbf + r * PROWB + c4;
    pb[0] = (__bf16)p.x; pb[1] = (__bf16)p.y; pb[2] = (__bf16)p.z; pb[3] = (__bf16)p.w;
  }
  __syncthreads();

  // ---- Phase 4: O = P V; wave w owns output cols [16w, 16w+16) --------------
  {
    const int n0 = wave * 16;
    const float* vcol = Vb + n0 + l16;           // B operand: col n = n0+l16
    const __bf16* prow = Pbf + l16 * PROWB;      // A operand: row m = l16
    v8f acc = {};
#pragma unroll 2
    for (int c = 0; c < S_ / 32; ++c) {
      v16bf a, b;
      const int ka = c * 32 + hi * 8;
#pragma unroll
      for (int e = 0; e < 8; ++e) { a[e] = prow[ka + e]; a[8 + e] = prow[ka + 16 + e]; }
      const int kb = c * 32 + hi * 8;
#pragma unroll
      for (int e = 0; e < 8; ++e) { b[e] = (__bf16)vcol[(size_t)(kb + e) * D_]; b[8 + e] = (__bf16)vcol[(size_t)(kb + 16 + e) * D_]; }
      acc = __builtin_amdgcn_wmma_f32_16x16x32_bf16(false, a, false, b,
                                                    (short)0, acc, false, false);
    }
    // O is write-once as well -> NT stores
#pragma unroll
    for (int i = 0; i < 8; ++i)
      __builtin_nontemporal_store(acc[i], Ob + (size_t)(i + hi * 8) * D_ + n0 + l16);
  }
}

extern "C" void kernel_launch(void* const* d_in, const int* in_sizes, int n_in,
                              void* d_out, int out_size, void* d_ws, size_t ws_size,
                              hipStream_t stream) {
  const float* Q = (const float*)d_in[0];
  const float* K = (const float*)d_in[1];
  const float* V = (const float*)d_in[2];
  float* Out  = (float*)d_out;
  float* Attw = Out + (size_t)B_ * H_ * S_ * D_;   // tuple order: (output, weights)

  const dim3 grid(B_ * H_ * (S_ / QT));            // 8192 workgroups
  const dim3 block(TPB);                           // 8 wave32 waves
  hipLaunchKernelGGL(ScaledDotProductAttention_29094108463701_kernel,
                     grid, block, SMEM_BYTES, stream, Q, K, V, Out, Attw);
}
